// RiemannianBatchNorm_31327491457695
// MI455X (gfx1250) — compile-verified, matrix operand order checked by reading, not run on hardware
//
#include <hip/hip_runtime.h>

// ---------------------------------------------------------------------------
// RiemannianBatchNorm on the Poincare ball, MI455X (gfx1250, wave32).
// N=16384 points, D=256, fp32. Latency/bandwidth bound; x is L2-resident.
// CDNA5 paths: TDM tensor_load_to_lds (double-buffered so DMA overlaps
// compute), WMMA f16->f32 ones-matmul for the big column-sum reductions
// (B operand staged column-major in LDS so it loads as 2x ds_load_b128),
// wave32 shfl butterflies. Deterministic: fixed-order reductions, no atomics.
// ---------------------------------------------------------------------------

#define EPSF       1e-6f
#define TANH_CLIPF (1.0f - 1e-5f)

typedef __attribute__((ext_vector_type(16))) _Float16 v16h;
typedef __attribute__((ext_vector_type(8)))  _Float16 v8h;
typedef __attribute__((ext_vector_type(8)))  float    v8f;

#if __has_builtin(__builtin_amdgcn_wmma_f32_16x16x32_f16)
#define HAVE_WMMA 1
#else
#define HAVE_WMMA 0
#endif

#if __has_builtin(__builtin_amdgcn_tensor_load_to_lds) && \
    __has_builtin(__builtin_amdgcn_s_wait_tensorcnt)
#define HAVE_TDM 1
#else
#define HAVE_TDM 0
#endif

// ------------------------------- helpers -----------------------------------

__device__ __forceinline__ float wsum(float v) {
#pragma unroll
  for (int off = 16; off; off >>= 1) v += __shfl_xor(v, off, 32);
  return v;
}

__device__ __forceinline__ float block_sum256(float v, float* sb) {
  const int t = threadIdx.x;
  sb[t] = v;
  __syncthreads();
#pragma unroll
  for (int s = 128; s > 0; s >>= 1) {
    if (t < s) sb[t] += sb[t + s];
    __syncthreads();
  }
  float r = sb[0];
  __syncthreads();
  return r;
}

// Lane `lane` owns columns [4*lane,4*lane+3] and [128+4*lane, 128+4*lane+3].
__device__ __forceinline__ void load8(const float* base, int lane, float* r) {
  const float4* v = (const float4*)base;
  float4 a = v[lane], b = v[lane + 32];
  r[0]=a.x; r[1]=a.y; r[2]=a.z; r[3]=a.w;
  r[4]=b.x; r[5]=b.y; r[6]=b.z; r[7]=b.w;
}
__device__ __forceinline__ void store8(float* base, int lane, const float* r) {
  float4* v = (float4*)base;
  v[lane]      = make_float4(r[0], r[1], r[2], r[3]);
  v[lane + 32] = make_float4(r[4], r[5], r[6], r[7]);
}

// o = mobius_add(a, b); a2 = ||a||^2 (precomputed, wave-uniform).
// Does the b.b and a.b wave reductions internally.
__device__ __forceinline__ void madd8(const float* a, float a2, const float* b,
                                      float c, float* o) {
  float b2 = 0.f, ab = 0.f;
#pragma unroll
  for (int j = 0; j < 8; ++j) { b2 += b[j]*b[j]; ab += a[j]*b[j]; }
  b2 = wsum(b2); ab = wsum(ab);
  float den = fmaxf(1.f + 2.f*c*ab + c*c*a2*b2, EPSF);
  float ca  = (1.f + 2.f*c*ab + c*b2) / den;
  float cb  = (1.f - c*a2) / den;
#pragma unroll
  for (int j = 0; j < 8; ++j) o[j] = ca*a[j] + cb*b[j];
}

// ------------------------------- TDM ---------------------------------------

#if HAVE_TDM
typedef unsigned int tdm_u32x4 __attribute__((ext_vector_type(4)));
typedef int          tdm_i32x8 __attribute__((ext_vector_type(8)));
typedef int          tdm_i32x4 __attribute__((ext_vector_type(4)));

// DMA a contiguous 8192-element (32KB) fp32 tile from global to LDS.
// 1D tensor: tensor_dim0 = tile_dim0 = 8192, data_size = 4B, type = 2.
__device__ __forceinline__ void tdm_load_tile_8k(const float* gsrc,
                                                 unsigned lds_off) {
  unsigned long long ga = (unsigned long long)(size_t)gsrc;
  tdm_u32x4 g0;
  g0[0] = 1u;                                               // count=1 (user)
  g0[1] = lds_off;                                          // lds_addr
  g0[2] = (unsigned)ga;                                     // global_addr lo
  g0[3] = (unsigned)((ga >> 32) & 0x01FFFFFFull) | (2u << 30); // hi + type=2
  tdm_i32x8 g1;
  g1[0] = (int)(2u << 16);        // workgroup_mask=0, data_size=2 (4 bytes)
  g1[1] = (int)(0x2000u << 16);   // tensor_dim0 lo16 = 8192
  g1[2] = (int)(1u << 16);        // tensor_dim0 hi16 = 0, tensor_dim1 = 1
  g1[3] = (int)(0x2000u << 16);   // tensor_dim1 hi16 = 0, tile_dim0 = 8192
  g1[4] = 0;                      // tile_dim1 = 0 (unused), tile_dim2 = 0
  g1[5] = 8192;                   // tensor_dim0_stride lo32
  g1[6] = 0;
  g1[7] = 0;
  tdm_i32x4 z4 = {0, 0, 0, 0};
#if __clang_major__ >= 23
  tdm_i32x8 z8 = {0, 0, 0, 0, 0, 0, 0, 0};
  __builtin_amdgcn_tensor_load_to_lds(g0, g1, z4, z4, z8, 0);
#else
  __builtin_amdgcn_tensor_load_to_lds(g0, g1, z4, z4, 0);
#endif
}
#endif

// ------------------------------- kernels -----------------------------------

// on_manifold = exp0(mean_param, c).  <<<1,256>>>
__global__ void k_exp0(const float* __restrict__ meanp,
                       const float* __restrict__ Kp,
                       float* __restrict__ om) {
  __shared__ float sb[256];
  const float c = -Kp[0], sc = sqrtf(c);
  const int t = threadIdx.x;
  float u   = meanp[t];
  float un2 = block_sum256(u * u, sb);
  float un  = sqrtf(fmaxf(un2, EPSF));
  om[t] = tanhf(sc * un) * u / (sc * un);
}

// Partial column sums of x (for the initial Euclidean mean).  <<<64,256>>>
__global__ void k_colsum(const float* __restrict__ x, float* __restrict__ part,
                         int rows_per_blk) {
  const int t = threadIdx.x;
  const size_t base = (size_t)blockIdx.x * rows_per_blk;
  float acc = 0.f;
  for (int r = 0; r < rows_per_blk; ++r) {
    __builtin_prefetch(x + (base + r + 8) * 256 + t, 0, 0);
    acc += x[(base + r) * 256 + t];
  }
  part[blockIdx.x * 256 + t] = acc;
}

// mu = clip_to_ball(mean(x)).  <<<1,256>>>
__global__ void k_mu_init(const float* __restrict__ part,
                          const float* __restrict__ Kp,
                          float* __restrict__ mu, int nblk, int N) {
  __shared__ float sb[256];
  const int t = threadIdx.x;
  const float c = -Kp[0], sc = sqrtf(c);
  float m = 0.f;
  for (int b = 0; b < nblk; ++b) m += part[b * 256 + t];
  m /= (float)N;
  float x2 = block_sum256(m * m, sb);
  float mn = sqrtf(fmaxf(x2, EPSF));
  float maxn = (1.0f - 1e-3f) / sc;
  mu[t] = m * fminf(1.f, maxn / mn);
}

// One Frechet pass: partial column sums of logmap(mu, x_i).  <<<128,256>>>
// TDM double-buffers 32x256 fp32 tiles into LDS (tile t+1's DMA overlaps
// tile t's compute); 8 waves compute 4 points each; l written column-major
// to LDS as f16; WMMA ones-matmul reduces 32 points x 16 columns per
// instruction into f32 accumulators, B operand = 2x ds_load_b128 per lane.
__global__ void __launch_bounds__(256)
k_iter(const float* __restrict__ x, const float* __restrict__ muP,
       const float* __restrict__ Kp, float* __restrict__ part, int ntiles) {
  __shared__ float    xtile[2][32][256];  // 64 KB, double-buffered TDM dest
  __shared__ _Float16 lbuf[256][40];      // 20 KB, column-major (pad->80B row)
  const int tid  = threadIdx.x;
  const int lane = tid & 31;
  const int w    = tid >> 5;
  const float c = -Kp[0], sc = sqrtf(c);

  float mu8[8], nmu8[8];
  load8(muP, lane, mu8);
  float x2mu = 0.f;
#pragma unroll
  for (int j = 0; j < 8; ++j) { x2mu += mu8[j]*mu8[j]; nmu8[j] = -mu8[j]; }
  x2mu = wsum(x2mu);
  const float lam   = 2.f / fmaxf(1.f - c * x2mu, EPSF);
  const float lcoef = 2.f / (sc * lam);

#if HAVE_WMMA
  v16h ones;
#pragma unroll
  for (int i = 0; i < 16; ++i) ones[i] = (_Float16)1.0f;
  v8f cacc0 = {0, 0, 0, 0, 0, 0, 0, 0};
  v8f cacc1 = {0, 0, 0, 0, 0, 0, 0, 0};
#else
  float sacc = 0.f;
#endif

  const float* xblk = x + (size_t)blockIdx.x * ntiles * 32 * 256;

#if HAVE_TDM
  if (tid < 32)  // wave 0 primes the pipeline with tile 0
    tdm_load_tile_8k(xblk, (unsigned)(size_t)&xtile[0][0][0]);
#endif

  for (int t = 0; t < ntiles; ++t) {
#if HAVE_TDM
    if (tid < 32) __builtin_amdgcn_s_wait_tensorcnt(0);   // tile t landed
#else
    {
      const float4* s4 = (const float4*)(xblk + (size_t)t * 32 * 256);
      float4* d4 = (float4*)&xtile[t & 1][0][0];
      for (int e = tid; e < 32 * 256 / 4; e += 256) d4[e] = s4[e];
    }
#endif
    __syncthreads();  // publish tile t; also orders last tile's lbuf reads

#if HAVE_TDM
    if (tid < 32 && t + 1 < ntiles)  // overlap tile t+1 DMA with compute
      tdm_load_tile_8k(xblk + (size_t)(t + 1) * 32 * 256,
                       (unsigned)(size_t)&xtile[(t + 1) & 1][0][0]);
#endif

    // logmap(mu, x_i) for this wave's 4 points -> column-major f16 LDS
#pragma unroll
    for (int q = 0; q < 4; ++q) {
      const int p = w * 4 + q;
      float y[8];
      load8(&xtile[t & 1][p][0], lane, y);
      float m[8];
      madd8(nmu8, x2mu, y, c, m);               // mobius_add(-mu, y)
      float mn2 = 0.f;
#pragma unroll
      for (int j = 0; j < 8; ++j) mn2 += m[j]*m[j];
      mn2 = wsum(mn2);
      float mn = sqrtf(fmaxf(mn2, EPSF));
      float tt = atanhf(fminf(sc * mn, TANH_CLIPF));
      float cf = lcoef * tt / mn;
#pragma unroll
      for (int j = 0; j < 4; ++j) {
        lbuf[lane * 4 + j][p]       = (_Float16)(cf * m[j]);
        lbuf[128 + lane * 4 + j][p] = (_Float16)(cf * m[4 + j]);
      }
    }
    __syncthreads();

#if HAVE_WMMA
    {
      // B = 32x16 f16 chunk: lanes 0..15 carry K=0..15, lanes 16..31 carry
      // K=16..31 -> one contiguous 16-half run per lane = 2x ds_load_b128.
      const int n  = lane & 15;
      const int kb = (lane >> 4) * 16;
#pragma unroll
      for (int cg = 0; cg < 2; ++cg) {
        const int col = w * 32 + cg * 16 + n;
        const v8h* rp = (const v8h*)&lbuf[col][kb];
        v8h lo = rp[0], hi = rp[1];
        v16h b;
#pragma unroll
        for (int i2 = 0; i2 < 8; ++i2) { b[i2] = lo[i2]; b[8 + i2] = hi[i2]; }
        if (cg == 0)
          cacc0 = __builtin_amdgcn_wmma_f32_16x16x32_f16(
              false, ones, false, b, (short)0, cacc0, false, false);
        else
          cacc1 = __builtin_amdgcn_wmma_f32_16x16x32_f16(
              false, ones, false, b, (short)0, cacc1, false, false);
      }
    }
#else
    {
      float s = 0.f;
      for (int p = 0; p < 32; ++p) s += (float)lbuf[tid][p];
      sacc += s;
    }
#endif
  }

#if HAVE_WMMA
  // C row 0 lives in VGPR0 of lanes 0..15.
  if (lane < 16) {
    part[blockIdx.x * 256 + w * 32 + lane]      = cacc0[0];
    part[blockIdx.x * 256 + w * 32 + 16 + lane] = cacc1[0];
  }
#else
  part[blockIdx.x * 256 + tid] = sacc;
#endif
}

// mu <- expmap(mu, mean_l).  <<<1,256>>>
__global__ void k_mu_update(const float* __restrict__ part,
                            const float* __restrict__ Kp,
                            float* __restrict__ mu, int nblk, int N) {
  __shared__ float sb[256];
  const int t = threadIdx.x;
  const float c = -Kp[0], sc = sqrtf(c);
  float g = 0.f;
  for (int b = 0; b < nblk; ++b) g += part[b * 256 + t];
  g /= (float)N;
  float m   = mu[t];
  float x2  = block_sum256(m * m, sb);
  float lam = 2.f / fmaxf(1.f - c * x2, EPSF);
  float un2 = block_sum256(g * g, sb);
  float un  = sqrtf(fmaxf(un2, EPSF));
  float sec = tanhf(sc * lam * un * 0.5f) * g / (sc * un);
  float y2  = block_sum256(sec * sec, sb);
  float xy  = block_sum256(m * sec, sb);
  float den = fmaxf(1.f + 2.f*c*xy + c*c*x2*y2, EPSF);
  float ax  = (1.f + 2.f*c*xy + c*y2) / den;
  float ay  = (1.f - c*x2) / den;
  mu[t] = ax * m + ay * sec;
}

// Partial sums of dist(x_i, mu)^2.  <<<128,256>>>, wave per point.
__global__ void __launch_bounds__(256)
k_var(const float* __restrict__ x, const float* __restrict__ muP,
      const float* __restrict__ Kp, float* __restrict__ vpart, int ppw) {
  __shared__ float sb[8];
  const int lane = threadIdx.x & 31, w = threadIdx.x >> 5;
  const float c = -Kp[0], sc = sqrtf(c);
  float mu8[8], nmu8[8];
  load8(muP, lane, mu8);
  float x2mu = 0.f;
#pragma unroll
  for (int j = 0; j < 8; ++j) { x2mu += mu8[j]*mu8[j]; nmu8[j] = -mu8[j]; }
  x2mu = wsum(x2mu);
  float acc = 0.f;
  for (int q = 0; q < ppw; ++q) {
    const size_t i = (size_t)blockIdx.x * (8 * ppw) + (size_t)w * ppw + q;
    __builtin_prefetch(x + (i + 1) * 256, 0, 0);
    float y[8];
    load8(x + i * 256, lane, y);
    float m[8];
    madd8(nmu8, x2mu, y, c, m);
    float mn2 = 0.f;
#pragma unroll
    for (int j = 0; j < 8; ++j) mn2 += m[j]*m[j];
    mn2 = wsum(mn2);
    float mn = sqrtf(fmaxf(mn2, EPSF));
    float d  = (2.f / sc) * atanhf(fminf(sc * mn, TANH_CLIPF));
    acc += d * d;
  }
  if (lane == 0) sb[w] = acc;
  __syncthreads();
  if (threadIdx.x == 0) {
    float s = 0.f;
    for (int j = 0; j < 8; ++j) s += sb[j];
    vpart[blockIdx.x] = s;
  }
}

// var -> scale; precompute p = mobius_add(om, -mu).  <<<1,256>>>
__global__ void k_var_fin(const float* __restrict__ vpart,
                          const float* __restrict__ varP,
                          const float* __restrict__ Kp,
                          const float* __restrict__ mu,
                          const float* __restrict__ om,
                          float* __restrict__ pv, float* __restrict__ scal,
                          int nblk, int N) {
  __shared__ float sb[256];
  const int t = threadIdx.x;
  const float c = -Kp[0];
  float v   = (t < nblk) ? vpart[t] : 0.f;
  float var = block_sum256(v, sb) / (float)N;
  float a = om[t], b = -mu[t];
  float x2 = block_sum256(a * a, sb);
  float y2 = block_sum256(b * b, sb);
  float xy = block_sum256(a * b, sb);
  float den = fmaxf(1.f + 2.f*c*xy + c*c*x2*y2, EPSF);
  pv[t] = ((1.f + 2.f*c*xy + c*y2) * a + (1.f - c*x2) * b) / den;
  if (t == 0) scal[0] = sqrtf(varP[0] / (var + 1e-6f));
}

// out = expmap(om, scale * transp(mu->om, logmap(mu, x))).  <<<128,256>>>
__global__ void __launch_bounds__(256)
k_final(const float* __restrict__ x, const float* __restrict__ muP,
        const float* __restrict__ omP, const float* __restrict__ pP,
        const float* __restrict__ scal, const float* __restrict__ Kp,
        float* __restrict__ out, int ppw) {
  const int lane = threadIdx.x & 31, w = threadIdx.x >> 5;
  const float c = -Kp[0], sc = sqrtf(c);
  float mu8[8], om8[8], p8[8], nmu8[8], np8[8];
  load8(muP, lane, mu8);
  load8(omP, lane, om8);
  load8(pP, lane, p8);
  float x2mu = 0.f, x2om = 0.f, x2p = 0.f;
#pragma unroll
  for (int j = 0; j < 8; ++j) {
    x2mu += mu8[j]*mu8[j]; x2om += om8[j]*om8[j]; x2p += p8[j]*p8[j];
    nmu8[j] = -mu8[j]; np8[j] = -p8[j];
  }
  x2mu = wsum(x2mu); x2om = wsum(x2om); x2p = wsum(x2p);
  const float lam_mu = 2.f / fmaxf(1.f - c * x2mu, EPSF);
  const float lam_om = 2.f / fmaxf(1.f - c * x2om, EPSF);
  const float fac    = scal[0] * lam_mu / lam_om;
  const float lcoef  = 2.f / (sc * lam_mu);

  for (int q = 0; q < ppw; ++q) {
    const size_t i = (size_t)blockIdx.x * (8 * ppw) + (size_t)w * ppw + q;
    __builtin_prefetch(x + (i + 1) * 256, 0, 0);
    float y[8];
    load8(x + i * 256, lane, y);

    // u = logmap(mu, y)
    float m[8];
    madd8(nmu8, x2mu, y, c, m);
    float mn2 = 0.f;
#pragma unroll
    for (int j = 0; j < 8; ++j) mn2 += m[j]*m[j];
    mn2 = wsum(mn2);
    float mn = sqrtf(fmaxf(mn2, EPSF));
    float cf = lcoef * atanhf(fminf(sc * mn, TANH_CLIPF)) / mn;
    float u8[8];
#pragma unroll
    for (int j = 0; j < 8; ++j) u8[j] = cf * m[j];

    // gyr[om, -mu] u = -(om(+)-mu) (+) (om (+) (-mu (+) u)), p = om(+)-mu
    float t1[8]; madd8(nmu8, x2mu, u8, c, t1);
    float t2[8]; madd8(om8, x2om, t1, c, t2);
    float t3[8]; madd8(np8, x2p, t2, c, t3);

    // scaled transported log, then expmap at om
    float lm[8];
#pragma unroll
    for (int j = 0; j < 8; ++j) lm[j] = fac * t3[j];
    float un2 = 0.f;
#pragma unroll
    for (int j = 0; j < 8; ++j) un2 += lm[j]*lm[j];
    un2 = wsum(un2);
    float un = sqrtf(fmaxf(un2, EPSF));
    float sf = tanhf(sc * lam_om * un * 0.5f) / (sc * un);
    float sec[8];
#pragma unroll
    for (int j = 0; j < 8; ++j) sec[j] = sf * lm[j];
    float o8[8];
    madd8(om8, x2om, sec, c, o8);
    store8(out + i * 256, lane, o8);
  }
}

// ------------------------------- host --------------------------------------

extern "C" void kernel_launch(void* const* d_in, const int* in_sizes, int n_in,
                              void* d_out, int out_size, void* d_ws,
                              size_t ws_size, hipStream_t stream) {
  const float* x     = (const float*)d_in[0];
  const float* Kp    = (const float*)d_in[1];
  const float* meanp = (const float*)d_in[2];
  const float* varp  = (const float*)d_in[3];
  float* out = (float*)d_out;

  const int D = 256;
  const int N = in_sizes[0] / D;        // 16384

  float* ws       = (float*)d_ws;
  float* mu       = ws;                 // 256
  float* om       = ws + 256;           // 256
  float* pv       = ws + 512;           // 256
  float* scal     = ws + 768;           // 16
  float* partials = ws + 784;           // 128*256
  float* varpart  = partials + 128 * 256;  // 128

  k_exp0<<<1, 256, 0, stream>>>(meanp, Kp, om);
  k_colsum<<<64, 256, 0, stream>>>(x, partials, N / 64);
  k_mu_init<<<1, 256, 0, stream>>>(partials, Kp, mu, 64, N);

  const int ntiles = N / (128 * 32);    // 4 tiles of 32 points per block
  for (int it = 0; it < 10; ++it) {
    k_iter<<<128, 256, 0, stream>>>(x, mu, Kp, partials, ntiles);
    k_mu_update<<<1, 256, 0, stream>>>(partials, Kp, mu, 128, N);
  }

  const int ppw = N / (128 * 8);        // 16 points per wave
  k_var<<<128, 256, 0, stream>>>(x, mu, Kp, varpart, ppw);
  k_var_fin<<<1, 256, 0, stream>>>(varpart, varp, Kp, mu, om, pv, scal, 128, N);
  k_final<<<128, 256, 0, stream>>>(x, mu, om, pv, scal, Kp, out, ppw);
}
